// PAGTNSpectraRegressor_82265803588400
// MI455X (gfx1250) — matrix operand order repaired and checked by reading, not compiled, not run on hardware
//
#include <hip/hip_runtime.h>
#include <hip/hip_bf16.h>
#include <math.h>

#define DEVFN __device__ __forceinline__

typedef unsigned short bf16_t;                                   // raw bf16 bits
typedef __bf16 bf16x16 __attribute__((ext_vector_type(16)));
typedef float  f32x8   __attribute__((ext_vector_type(8)));

static const int Nn = 50000, Ee = 400000, Bb = 512;

DEVFN bf16_t f2bf(float f) {
  unsigned u = __float_as_uint(f);
  return (bf16_t)((u + 0x7FFFu + ((u >> 16) & 1u)) >> 16);       // RNE
}
DEVFN float bf2f(bf16_t b) { return __uint_as_float(((unsigned)b) << 16); }
DEVFN float lrelu(float v) { return v >= 0.f ? v : 0.2f * v; }
DEVFN float sigm(float x) { return 1.f / (1.f + expf(-x)); }
// order-preserving float<->uint mapping for atomic segment-max
DEVFN unsigned f2ord(float f) {
  unsigned u = __float_as_uint(f);
  return (u & 0x80000000u) ? ~u : (u | 0x80000000u);
}
DEVFN float ord2f(unsigned o) {
  unsigned u = (o & 0x80000000u) ? (o & 0x7FFFFFFFu) : ~o;
  return __uint_as_float(u);
}

DEVFN void store_out(bf16_t* p, float v) { *p = f2bf(v); }
DEVFN void store_out(float* p, float v) { *p = v; }

// ---------------------------------------------------------------------------
// WMMA GEMM: Y[M,Nc] = act(A[M,K](bf16) @ W[K,Nc](bf16) + bias)
// block = 256 threads = 8 waves; tile 128(M) x 64(N); K step 32.
// A tile staged with GLOBAL_LOAD_ASYNC_TO_LDS_B128 (ASYNCcnt-tracked DMA);
// B tile loaded coalesced b128 and transposed through LDS.
// Each wave owns a 16x64 strip -> 4 x v_wmma_f32_16x16x32_bf16 per K step.
// ACT: 0 none, 1 lrelu(0.2), 2 relu, 3 prelu(*alpha)
// ---------------------------------------------------------------------------
template <int ACT, typename OutT>
__global__ __launch_bounds__(256) void wmma_gemm_kernel(
    const bf16_t* __restrict__ A, const bf16_t* __restrict__ W,
    const float* __restrict__ bias, const float* __restrict__ alpha,
    OutT* __restrict__ Y, int M, int K, int Nc) {
  __shared__ bf16_t As[128][32];
  __shared__ bf16_t Bt[64][32];   // B tile transposed: Bt[n][k]

  const int tid = threadIdx.x;
  const int wave = tid >> 5, lane = tid & 31;
  const int m0 = blockIdx.x * 128;
  const int n0 = blockIdx.y * 64;

  f32x8 acc[4];
#pragma unroll
  for (int t = 0; t < 4; t++)
#pragma unroll
    for (int i = 0; i < 8; i++) acc[t][i] = 0.f;

  const int arow = tid >> 1;         // 0..127
  const int ahalf = (tid & 1) * 16;  // 0 / 16  (elements)
  const int brow = tid >> 3;         // 0..31   (k within tile)
  const int bj8 = (tid & 7) * 8;     // 0..56   (n within tile, x8)

  // LDS byte offset of this thread's A staging slot (ISA 10.2: generic LDS
  // address truncates to the LDS byte offset in its low 32 bits)
  const unsigned lds_a = (unsigned)(size_t)(void*)&As[arow][ahalf];

  for (int k0 = 0; k0 < K; k0 += 32) {
    // ---- stage A 128x32 bf16 tile: per-lane async DMA, 2 x 16B
    {
      int gr = m0 + arow;
      if (gr < M) {
        const bf16_t* gp = A + (size_t)gr * K + k0 + ahalf;
        asm volatile("global_load_async_to_lds_b128 %0, %1, off"
                     :: "v"(lds_a), "v"((unsigned long long)gp) : "memory");
        asm volatile("global_load_async_to_lds_b128 %0, %1, off offset:16"
                     :: "v"(lds_a), "v"((unsigned long long)gp) : "memory");
        if (k0 + 32 < K)  // gfx1250 global_prefetch_b8 for next K tile
          __builtin_prefetch((const void*)(gp + 32), 0, 1);
      } else {
        uint4 z = make_uint4(0u, 0u, 0u, 0u);
        *(uint4*)(&As[arow][ahalf]) = z;
        *(uint4*)(&As[arow][ahalf + 8]) = z;
      }
    }
    // ---- stage B transposed: coalesced b128 row load, ds_store_b16 scatter
    {
      union { uint4 q; bf16_t e[8]; } wv;
      wv.q = *(const uint4*)(W + (size_t)(k0 + brow) * Nc + n0 + bj8);
#pragma unroll
      for (int jj = 0; jj < 8; jj++) Bt[bj8 + jj][brow] = wv.e[jj];
    }
    asm volatile("s_wait_asynccnt 0x0" ::: "memory");  // own wave's DMA done
    __syncthreads();                                   // all waves' LDS visible

    // ---- A fragment: 16-bit A 16x32 layout (ISA 7.12.2):
    // lanes 0-15: elems 0..7 = K0..7, 8..15 = K16..23 (row M = lane)
    // lanes16-31: elems 0..7 = K8..15, 8..15 = K24..31
    union { bf16x16 v; uint4 q[2]; } af;
    {
      int r = wave * 16 + (lane & 15);
      int ka = (lane < 16) ? 0 : 8;
      af.q[0] = *(const uint4*)(&As[r][ka]);
      af.q[1] = *(const uint4*)(&As[r][ka + 16]);
    }
#pragma unroll
    for (int t = 0; t < 4; t++) {
      // B fragment: col N = lane%16, lanes<16 hold K0..7,K16..23, lanes>=16 K8..15,K24..31
      union { bf16x16 v; uint4 q[2]; } bfr;
      int n = t * 16 + (lane & 15);
      int kb = (lane < 16) ? 0 : 8;
      bfr.q[0] = *(const uint4*)(&Bt[n][kb]);
      bfr.q[1] = *(const uint4*)(&Bt[n][kb + 16]);
      acc[t] = __builtin_amdgcn_wmma_f32_16x16x32_bf16(
          false, af.v, false, bfr.v, (short)0, acc[t], false, false);
    }
    __syncthreads();
  }

  // ---- epilogue per C/D layout: VGPR r -> row r (lanes<16) / 8+r (lanes>=16)
  const float al = (ACT == 3) ? alpha[0] : 0.f;
  const int rowBase = m0 + wave * 16 + ((lane < 16) ? 0 : 8);
  const int colLane = lane & 15;
#pragma unroll
  for (int t = 0; t < 4; t++) {
    int col = n0 + t * 16 + colLane;
    float bc = bias ? bias[col] : 0.f;
#pragma unroll
    for (int r = 0; r < 8; r++) {
      int row = rowBase + r;
      if (row < M) {
        float v = acc[t][r] + bc;
        if (ACT == 1) v = lrelu(v);
        else if (ACT == 2) v = fmaxf(v, 0.f);
        else if (ACT == 3) v = (v >= 0.f) ? v : al * v;
        store_out(&Y[(size_t)row * Nc + col], v);
      }
    }
  }
}

// ---------------------------------------------------------------------------
// Elementwise / scatter kernels
// ---------------------------------------------------------------------------
__global__ void k_f32_to_bf16(const float* __restrict__ s, bf16_t* __restrict__ d, int n) {
  int i = blockIdx.x * blockDim.x + threadIdx.x;
  if (i < n) d[i] = f2bf(s[i]);
}
__global__ void k_transpose_bf(const float* __restrict__ s, bf16_t* __restrict__ d, int R, int C) {
  int i = blockIdx.x * blockDim.x + threadIdx.x;   // s:[R,C] -> d:[C,R]
  if (i < R * C) { int r = i / C, c = i % C; d[c * R + r] = f2bf(s[i]); }
}
__global__ void k_copy_bf(const bf16_t* __restrict__ s, bf16_t* __restrict__ d, int n) {
  int i = blockIdx.x * blockDim.x + threadIdx.x;
  if (i < n) d[i] = s[i];
}
__global__ void k_fill_f32(float* p, float v, int n) {
  int i = blockIdx.x * blockDim.x + threadIdx.x;
  if (i < n) p[i] = v;
}
__global__ void k_fill_u32(unsigned* p, unsigned v, int n) {
  int i = blockIdx.x * blockDim.x + threadIdx.x;
  if (i < n) p[i] = v;
}

// score[e,h] = dot(lrelu(as[src]+ad[dst]+ae[e]), dot_w) + dot_b   (warp per e*2+h)
__global__ void k_edge_score(const bf16_t* __restrict__ as_, const bf16_t* __restrict__ ad,
                             const bf16_t* __restrict__ ae, const float* __restrict__ dw,
                             const float* __restrict__ db, const int* __restrict__ src,
                             const int* __restrict__ dst, float* __restrict__ score, int E_) {
  int warp = (blockIdx.x * blockDim.x + threadIdx.x) >> 5;
  int lane = threadIdx.x & 31;
  if (warp >= E_ * 2) return;
  int e = warp >> 1, h = warp & 1;
  int s = src[e], d = dst[e];
  float acc = 0.f;
#pragma unroll
  for (int i = 0; i < 4; i++) {
    int c = lane + 32 * i;
    float v = bf2f(as_[((size_t)s * 2 + h) * 128 + c]) +
              bf2f(ad[((size_t)d * 2 + h) * 128 + c]) +
              bf2f(ae[(size_t)e * 128 + c]);
    acc += lrelu(v) * dw[c];
  }
  for (int o = 16; o > 0; o >>= 1) acc += __shfl_xor(acc, o, 32);
  if (lane == 0) score[warp] = acc + db[0];
}

__global__ void k_seg_max_eh(const float* __restrict__ sc, const int* __restrict__ dst,
                             unsigned* __restrict__ mx, int E_) {
  int i = blockIdx.x * blockDim.x + threadIdx.x;
  if (i >= E_ * 2) return;
  atomicMax(&mx[dst[i >> 1] * 2 + (i & 1)], f2ord(sc[i]));
}
__global__ void k_seg_exp_eh(float* __restrict__ sc, const int* __restrict__ dst,
                             const unsigned* __restrict__ mx, float* __restrict__ den, int E_) {
  int i = blockIdx.x * blockDim.x + threadIdx.x;
  if (i >= E_ * 2) return;
  int seg = dst[i >> 1] * 2 + (i & 1);
  float ex = expf(sc[i] - ord2f(mx[seg]));
  sc[i] = ex;
  atomicAdd(&den[seg], ex);
}

// agg[dst,h,c] += (ex/den) * lrelu(ms[src,h,c] + me[e,c])  -- block(256) per edge
__global__ void k_edge_agg(const float* __restrict__ ex, const float* __restrict__ den,
                           const bf16_t* __restrict__ ms, const bf16_t* __restrict__ me,
                           const int* __restrict__ src, const int* __restrict__ dst,
                           float* __restrict__ agg, int E_) {
  int e = blockIdx.x;
  if (e >= E_) return;
  int h = threadIdx.x >> 7, c = threadIdx.x & 127;
  int s = src[e], d = dst[e];
  float attn = ex[e * 2 + h] / den[d * 2 + h];
  float m = lrelu(bf2f(ms[((size_t)s * 2 + h) * 128 + c]) + bf2f(me[(size_t)e * 128 + c]));
  atomicAdd(&agg[((size_t)d * 2 + h) * 128 + c], attn * m);
}

// h = relu(lrelu(agg + wgt) + h0)
__global__ void k_node_update(const float* __restrict__ agg, const bf16_t* __restrict__ wgt,
                              const bf16_t* __restrict__ h0, bf16_t* __restrict__ h, int n) {
  int i = blockIdx.x * blockDim.x + threadIdx.x;
  if (i >= n) return;
  float v = lrelu(agg[i] + bf2f(wgt[i])) + bf2f(h0[i]);
  h[i] = f2bf(fmaxf(v, 0.f));
}

// node_emb[n, 0:64]=x ; [64:192]=head-mean(h)
__global__ void k_node_emb(const float* __restrict__ x, const bf16_t* __restrict__ h,
                           float* __restrict__ ne, int n) {
  int i = blockIdx.x * blockDim.x + threadIdx.x;
  if (i >= n) return;
  int nn = i / 192, c = i % 192;
  float v;
  if (c < 64) v = x[(size_t)nn * 64 + c];
  else {
    int cc = c - 64;
    v = 0.5f * (bf2f(h[(size_t)nn * 256 + cc]) + bf2f(h[(size_t)nn * 256 + 128 + cc]));
  }
  ne[i] = v;
}

__global__ void k_bias_sum(const float* a, const float* b, float* o, int n) {
  int i = blockIdx.x * blockDim.x + threadIdx.x;
  if (i < n) o[i] = a[i] + b[i];
}

// LSTM cell (torch gate order i,f,g,o), z = Z1 + Z2
__global__ void k_lstm_cell(const float* __restrict__ Z1, const float* __restrict__ Z2,
                            float* __restrict__ cS, float* __restrict__ hS, int n) {
  int i = blockIdx.x * blockDim.x + threadIdx.x;
  if (i >= n) return;
  int b = i / 192, j = i % 192;
  const float* z1 = Z1 + (size_t)b * 768;
  const float* z2 = Z2 + (size_t)b * 768;
  float zi = z1[j] + z2[j];
  float zf = z1[192 + j] + z2[192 + j];
  float zg = z1[384 + j] + z2[384 + j];
  float zo = z1[576 + j] + z2[576 + j];
  float cn = sigm(zf) * cS[i] + sigm(zi) * tanhf(zg);
  cS[i] = cn;
  hS[i] = sigm(zo) * tanhf(cn);
}

// e[n] = dot(node_emb[n], hS[batch[n]])   (warp per node, 192 dims)
__global__ void k_s2s_energy(const float* __restrict__ ne, const float* __restrict__ hS,
                             const int* __restrict__ batch, float* __restrict__ eN, int n) {
  int warp = (blockIdx.x * blockDim.x + threadIdx.x) >> 5;
  int lane = threadIdx.x & 31;
  if (warp >= n) return;
  const float* q = hS + (size_t)batch[warp] * 192;
  const float* xv = ne + (size_t)warp * 192;
  float acc = 0.f;
#pragma unroll
  for (int i = 0; i < 6; i++) { int c = lane + 32 * i; acc += xv[c] * q[c]; }
  for (int o = 16; o > 0; o >>= 1) acc += __shfl_xor(acc, o, 32);
  if (lane == 0) eN[warp] = acc;
}
__global__ void k_seg_max_node(const float* eN, const int* batch, unsigned* mx, int n) {
  int i = blockIdx.x * blockDim.x + threadIdx.x;
  if (i < n) atomicMax(&mx[batch[i]], f2ord(eN[i]));
}
__global__ void k_seg_exp_node(float* eN, const int* batch, const unsigned* mx, float* den, int n) {
  int i = blockIdx.x * blockDim.x + threadIdx.x;
  if (i >= n) return;
  float e = expf(eN[i] - ord2f(mx[batch[i]]));
  eN[i] = e;
  atomicAdd(&den[batch[i]], e);
}
__global__ void k_s2s_r(const float* __restrict__ exN, const float* __restrict__ den,
                        const float* __restrict__ ne, const int* __restrict__ batch,
                        float* __restrict__ rB, int n) {
  int i = blockIdx.x * blockDim.x + threadIdx.x;
  if (i >= n) return;
  int nn = i / 192, c = i % 192;
  int b = batch[nn];
  float a = exN[nn] / den[b];
  atomicAdd(&rB[(size_t)b * 192 + c], a * ne[i]);
}
__global__ void k_qstar(const float* hS, const float* rB, float* qs, int n) {
  int i = blockIdx.x * blockDim.x + threadIdx.x;
  if (i >= n) return;
  int b = i / 384, c = i % 384;
  qs[i] = (c < 192) ? hS[b * 192 + c] : rB[b * 192 + (c - 192)];
}
// feats[n] = concat(node_emb[n], q_star[batch[n]]) -> bf16
__global__ void k_feats(const float* __restrict__ ne, const float* __restrict__ qs,
                        const int* __restrict__ batch, bf16_t* __restrict__ f, int n) {
  int i = blockIdx.x * blockDim.x + threadIdx.x;
  if (i >= n) return;
  int nn = i / 576, c = i % 576;
  float v = (c < 192) ? ne[(size_t)nn * 192 + c] : qs[(size_t)batch[nn] * 384 + (c - 192)];
  f[i] = f2bf(v);
}
// out[n] = mask[n] ? dot(h3[n], w3)+b3 : 0   (warp per node, 512 dims)
__global__ void k_final(const bf16_t* __restrict__ h3, const float* __restrict__ w3,
                        const float* __restrict__ b3, const unsigned char* __restrict__ mask,
                        float* __restrict__ out, int n) {
  int warp = (blockIdx.x * blockDim.x + threadIdx.x) >> 5;
  int lane = threadIdx.x & 31;
  if (warp >= n) return;
  const bf16_t* hv = h3 + (size_t)warp * 512;
  float acc = 0.f;
#pragma unroll
  for (int i = 0; i < 16; i++) { int c = lane + 32 * i; acc += bf2f(hv[c]) * w3[c]; }
  for (int o = 16; o > 0; o >>= 1) acc += __shfl_xor(acc, o, 32);
  if (lane == 0) out[warp] = mask[warp] ? (acc + b3[0]) : 0.f;
}

// ---------------------------------------------------------------------------
// Host orchestration
// ---------------------------------------------------------------------------
#define CDIV(a, b) (((a) + (b) - 1) / (b))
#define L1D(k, n, ...) k<<<CDIV((n), 256), 256, 0, stream>>>(__VA_ARGS__)

template <int ACT, typename OutT>
static void gemm(const bf16_t* A, const bf16_t* W, const float* bias, const float* alpha,
                 OutT* Y, int M, int K, int Nc, hipStream_t stream) {
  dim3 g(CDIV(M, 128), Nc / 64);
  wmma_gemm_kernel<ACT, OutT><<<g, 256, 0, stream>>>(A, W, bias, alpha, Y, M, K, Nc);
}

extern "C" void kernel_launch(void* const* d_in, const int* in_sizes, int n_in,
                              void* d_out, int out_size, void* d_ws, size_t ws_size,
                              hipStream_t stream) {
  (void)in_sizes; (void)n_in; (void)out_size; (void)ws_size;
  // ---- input pytree flattening (dict keys sorted) ----
  const float* x  = (const float*)d_in[0];
  const float* ea = (const float*)d_in[1];
  const int* eidx = (const int*)d_in[2];
  const int* src = eidx, * dst = eidx + Ee;
  const int* batch = (const int*)d_in[3];
  const unsigned char* mask = (const unsigned char*)d_in[4];
  const float* inp_b = (const float*)d_in[5];
  const float* inp_w = (const float*)d_in[6];
  // layer l: base 7+l*14 : dot.b,dot.w,dst.b,dst.w,edg.b,edg.w,src.b,src.w,
  //                        medg.b,medg.w,msrc.b,msrc.w,wgt.b,wgt.w
  const float* lstm_bhh = (const float*)d_in[63];
  const float* lstm_bih = (const float*)d_in[64];
  const float* lstm_whh = (const float*)d_in[65];
  const float* lstm_wih = (const float*)d_in[66];
  const float* m_b[4] = { (const float*)d_in[67], (const float*)d_in[69],
                          (const float*)d_in[71], (const float*)d_in[73] };
  const float* m_w[4] = { (const float*)d_in[68], (const float*)d_in[70],
                          (const float*)d_in[72], (const float*)d_in[74] };
  const float* pr[3] = { (const float*)d_in[75], (const float*)d_in[76], (const float*)d_in[77] };

  // ---- workspace bump allocator ----
  size_t off = 0;
  auto alloc = [&](size_t bytes) -> void* {
    void* p = (char*)d_ws + off;
    off += (bytes + 255) & ~(size_t)255;
    return p;
  };
  bf16_t* w_inp = (bf16_t*)alloc(64 * 256 * 2);
  bf16_t *w_dst[4], *w_edg[4], *w_src[4], *w_medg[4], *w_msrc[4], *w_wgt[4];
  for (int l = 0; l < 4; l++) {
    w_dst[l] = (bf16_t*)alloc(128 * 128 * 2); w_edg[l] = (bf16_t*)alloc(32 * 128 * 2);
    w_src[l] = (bf16_t*)alloc(128 * 128 * 2); w_medg[l] = (bf16_t*)alloc(32 * 128 * 2);
    w_msrc[l] = (bf16_t*)alloc(128 * 128 * 2); w_wgt[l] = (bf16_t*)alloc(128 * 128 * 2);
  }
  bf16_t* wihT = (bf16_t*)alloc((size_t)384 * 768 * 2);
  bf16_t* whhT = (bf16_t*)alloc((size_t)192 * 768 * 2);
  bf16_t* wm0 = (bf16_t*)alloc((size_t)576 * 512 * 2);
  bf16_t* wm1 = (bf16_t*)alloc((size_t)512 * 512 * 2);
  bf16_t* wm2 = (bf16_t*)alloc((size_t)512 * 512 * 2);
  bf16_t* x_bf = (bf16_t*)alloc((size_t)Nn * 64 * 2);
  bf16_t* ea_bf = (bf16_t*)alloc((size_t)Ee * 32 * 2);
  bf16_t* h0 = (bf16_t*)alloc((size_t)Nn * 256 * 2);
  bf16_t* h = (bf16_t*)alloc((size_t)Nn * 256 * 2);
  bf16_t* asb = (bf16_t*)alloc((size_t)Nn * 256 * 2);   // as_ , later reused for ms
  bf16_t* adb = (bf16_t*)alloc((size_t)Nn * 256 * 2);   // ad  , later reused for wgt
  bf16_t* emb = (bf16_t*)alloc((size_t)Ee * 128 * 2);   // ae  , later reused for me
  float* score = (float*)alloc((size_t)Ee * 2 * 4);
  unsigned* mxu = (unsigned*)alloc((size_t)Nn * 2 * 4);
  float* den = (float*)alloc((size_t)Nn * 2 * 4);
  float* agg = (float*)alloc((size_t)Nn * 256 * 4);
  float* ne = (float*)alloc((size_t)Nn * 192 * 4);
  float* bsum = (float*)alloc(768 * 4);
  float* qstar = (float*)alloc((size_t)Bb * 384 * 4);
  float* hS = (float*)alloc((size_t)Bb * 192 * 4);
  float* cS = (float*)alloc((size_t)Bb * 192 * 4);
  bf16_t* qs_bf = (bf16_t*)alloc((size_t)Bb * 384 * 2);
  bf16_t* hs_bf = (bf16_t*)alloc((size_t)Bb * 192 * 2);
  float* Z1 = (float*)alloc((size_t)Bb * 768 * 4);
  float* Z2 = (float*)alloc((size_t)Bb * 768 * 4);
  float* eN = (float*)alloc((size_t)Nn * 4);
  unsigned* mxB = (unsigned*)alloc(Bb * 4);
  float* denB = (float*)alloc(Bb * 4);
  float* rB = (float*)alloc((size_t)Bb * 192 * 4);
  bf16_t* feats = (bf16_t*)alloc((size_t)Nn * 576 * 2);
  bf16_t* h1 = (bf16_t*)alloc((size_t)Nn * 512 * 2);    // also final MLP layer out
  bf16_t* h2 = (bf16_t*)alloc((size_t)Nn * 512 * 2);

  // ---- weight conversion ----
  L1D(k_f32_to_bf16, 64 * 256, inp_w, w_inp, 64 * 256);
  for (int l = 0; l < 4; l++) {
    const int B0 = 7 + l * 14;
    L1D(k_f32_to_bf16, 128 * 128, (const float*)d_in[B0 + 3], w_dst[l], 128 * 128);
    L1D(k_f32_to_bf16, 32 * 128, (const float*)d_in[B0 + 5], w_edg[l], 32 * 128);
    L1D(k_f32_to_bf16, 128 * 128, (const float*)d_in[B0 + 7], w_src[l], 128 * 128);
    L1D(k_f32_to_bf16, 32 * 128, (const float*)d_in[B0 + 9], w_medg[l], 32 * 128);
    L1D(k_f32_to_bf16, 128 * 128, (const float*)d_in[B0 + 11], w_msrc[l], 128 * 128);
    L1D(k_f32_to_bf16, 128 * 128, (const float*)d_in[B0 + 13], w_wgt[l], 128 * 128);
  }
  L1D(k_transpose_bf, 768 * 384, lstm_wih, wihT, 768, 384);  // -> [384,768]
  L1D(k_transpose_bf, 768 * 192, lstm_whh, whhT, 768, 192);  // -> [192,768]
  L1D(k_f32_to_bf16, 576 * 512, m_w[0], wm0, 576 * 512);
  L1D(k_f32_to_bf16, 512 * 512, m_w[1], wm1, 512 * 512);
  L1D(k_f32_to_bf16, 512 * 512, m_w[2], wm2, 512 * 512);
  L1D(k_f32_to_bf16, Nn * 64, x, x_bf, Nn * 64);
  L1D(k_f32_to_bf16, Ee * 32, ea, ea_bf, Ee * 32);
  L1D(k_bias_sum, 768, lstm_bih, lstm_bhh, bsum, 768);

  // ---- input projection: h0 = lrelu(x @ Winp + b) ----
  gemm<1>(x_bf, w_inp, inp_b, nullptr, h0, Nn, 64, 256, stream);
  L1D(k_copy_bf, Nn * 256, h0, h, Nn * 256);

  // ---- 4 GNN layers ----
  for (int l = 0; l < 4; l++) {
    const int B0 = 7 + l * 14;
    const float* dot_b = (const float*)d_in[B0 + 0];
    const float* dot_w = (const float*)d_in[B0 + 1];
    // attention linears (act=none: lrelu applies after the 3-way sum)
    gemm<0>(h, w_src[l], (const float*)d_in[B0 + 6], nullptr, asb, Nn * 2, 128, 128, stream);
    gemm<0>(h, w_dst[l], (const float*)d_in[B0 + 2], nullptr, adb, Nn * 2, 128, 128, stream);
    gemm<0>(ea_bf, w_edg[l], (const float*)d_in[B0 + 4], nullptr, emb, Ee, 32, 128, stream);
    k_edge_score<<<CDIV(Ee * 2 * 32, 256), 256, 0, stream>>>(asb, adb, emb, dot_w, dot_b,
                                                             src, dst, score, Ee);
    L1D(k_fill_u32, Nn * 2, mxu, 0u, Nn * 2);
    L1D(k_fill_f32, Nn * 2, den, 0.f, Nn * 2);
    L1D(k_seg_max_eh, Ee * 2, score, dst, mxu, Ee);
    L1D(k_seg_exp_eh, Ee * 2, score, dst, mxu, den, Ee);
    // messages (reuse asb for ms, emb for me)
    gemm<0>(h, w_msrc[l], (const float*)d_in[B0 + 10], nullptr, asb, Nn * 2, 128, 128, stream);
    gemm<0>(ea_bf, w_medg[l], (const float*)d_in[B0 + 8], nullptr, emb, Ee, 32, 128, stream);
    L1D(k_fill_f32, Nn * 256, agg, 0.f, Nn * 256);
    k_edge_agg<<<Ee, 256, 0, stream>>>(score, den, asb, emb, src, dst, agg, Ee);
    // node update (reuse adb for wgt)
    gemm<0>(h, w_wgt[l], (const float*)d_in[B0 + 12], nullptr, adb, Nn * 2, 128, 128, stream);
    L1D(k_node_update, Nn * 256, agg, adb, h0, h, Nn * 256);
  }
  L1D(k_node_emb, Nn * 192, x, h, ne, Nn * 192);

  // ---- Set2Set (3 steps) ----
  L1D(k_fill_f32, Bb * 384, qstar, 0.f, Bb * 384);
  L1D(k_fill_f32, Bb * 192, hS, 0.f, Bb * 192);
  L1D(k_fill_f32, Bb * 192, cS, 0.f, Bb * 192);
  for (int it = 0; it < 3; it++) {
    L1D(k_f32_to_bf16, Bb * 384, qstar, qs_bf, Bb * 384);
    L1D(k_f32_to_bf16, Bb * 192, hS, hs_bf, Bb * 192);
    gemm<0>(qs_bf, wihT, bsum, nullptr, Z1, Bb, 384, 768, stream);
    gemm<0>(hs_bf, whhT, (const float*)nullptr, nullptr, Z2, Bb, 192, 768, stream);
    L1D(k_lstm_cell, Bb * 192, Z1, Z2, cS, hS, Bb * 192);
    k_s2s_energy<<<CDIV(Nn * 32, 256), 256, 0, stream>>>(ne, hS, batch, eN, Nn);
    L1D(k_fill_u32, Bb, mxB, 0u, Bb);
    L1D(k_fill_f32, Bb, denB, 0.f, Bb);
    L1D(k_seg_max_node, Nn, eN, batch, mxB, Nn);
    L1D(k_seg_exp_node, Nn, eN, batch, mxB, denB, Nn);
    L1D(k_fill_f32, Bb * 192, rB, 0.f, Bb * 192);
    L1D(k_s2s_r, Nn * 192, eN, denB, ne, batch, rB, Nn * 192);
    L1D(k_qstar, Bb * 384, hS, rB, qstar, Bb * 384);
  }

  // ---- MLP head ----
  L1D(k_feats, Nn * 576, ne, qstar, batch, feats, Nn * 576);
  gemm<3>(feats, wm0, m_b[0], pr[0], h1, Nn, 576, 512, stream);
  gemm<3>(h1, wm1, m_b[1], pr[1], h2, Nn, 512, 512, stream);
  gemm<3>(h2, wm2, m_b[2], pr[2], h1, Nn, 512, 512, stream);
  k_final<<<CDIV(Nn * 32, 256), 256, 0, stream>>>(h1, m_w[3], m_b[3], mask,
                                                  (float*)d_out, Nn);
}
